// TemporalCfCEncoder_90615220011699
// MI455X (gfx1250) — matrix operand order repaired and checked by reading, not compiled, not run on hardware
//
#include <hip/hip_runtime.h>
#include <cstdint>
#include <cstddef>

#define T_STEPS 512
#define K_TOK   256
#define DMODEL  256
#define NHEAD   4
#define DHEAD   64
#define NLAYER  2

typedef __attribute__((ext_vector_type(16))) _Float16 v16h;
typedef __attribute__((ext_vector_type(8)))  float    v8f;
typedef __attribute__((ext_vector_type(4)))  unsigned int v4u;
typedef __attribute__((ext_vector_type(8)))  int      v8i;
typedef __attribute__((ext_vector_type(4)))  int      v4i;
typedef _Float16 h16_t;

// Scheduling pipeline hints: 0x20=VMEM read, 0x8=MFMA/WMMA, 0x100=DS read,
// 0x200=DS write.
#if __has_builtin(__builtin_amdgcn_sched_group_barrier)
#define SGB(mask, cnt) __builtin_amdgcn_sched_group_barrier(mask, cnt, 0)
#else
#define SGB(mask, cnt)
#endif

// ---------------------------------------------------------------------------
// WMMA helpers (CDNA5 16x16x32 f16 -> f32)
// ---------------------------------------------------------------------------
__device__ __forceinline__ v8f wmma_f16(v16h a, v16h b, v8f c) {
  return __builtin_amdgcn_wmma_f32_16x16x32_f16(false, a, false, b, (short)0, c,
                                                false, false);
}

// A fragment: 16x32 f16 tile, row-major source with leading dim lda (halfs).
__device__ __forceinline__ v16h load_a_frag(const h16_t* p, int lda, int k0) {
  int l = threadIdx.x & 31;
  int row = l & 15, hk = (l >> 4) & 1;
  const h16_t* r0 = p + (size_t)row * lda + k0 + hk * 16;
  v16h a;
#pragma unroll
  for (int e = 0; e < 8; ++e) { a[e] = r0[e]; a[e + 8] = r0[8 + e]; }
  return a;
}

// B fragment: 32x16 tile of B, memory holds B^T row-major [N][K], ld=ldk.
__device__ __forceinline__ v16h load_bT_frag(const h16_t* p, int ldk, int k0) {
  int l = threadIdx.x & 31;
  int col = l & 15, kh = (l >> 4) & 1;
  const h16_t* r0 = p + (size_t)col * ldk + k0 + kh * 16;
  v16h b;
#pragma unroll
  for (int e = 0; e < 16; ++e) b[e] = r0[e];
  return b;
}

// ---------------------------------------------------------------------------
// TDM: async 2-D tile load of [16 x 256] f16 into LDS with 16B padding after
// each 512B row (LDS row stride 264 halfs). CDNA5 ISA.
// ---------------------------------------------------------------------------
__device__ __forceinline__ void tdm_load_tile16x256_f16(uint32_t lds_addr,
                                                        uint64_t gaddr) {
  v4u g0;
  g0[0] = 1u;                                         // count=1 (valid user D#)
  g0[1] = lds_addr;                                   // lds_addr
  g0[2] = (uint32_t)(gaddr & 0xFFFFFFFFu);            // global_addr[31:0]
  g0[3] = (uint32_t)((gaddr >> 32) & 0x1FFFFFFu)      // global_addr[56:32]
          | (2u << 30);                               // type=2 ("image")
  v8i g1;
  g1[0] = (int)((1u << 16)      // data_size = 2 bytes
                | (1u << 20)    // pad_enable
                | (6u << 22)    // pad_interval: 128 DWORDs (=512B row)
                | (3u << 25));  // pad_amount: 4 DWORDs (=16B)
  g1[1] = (int)(256u << 16);    // tensor_dim0 = 256
  g1[2] = 0;
  g1[3] = (int)(2u | (256u << 16));  // tensor_dim1 hi16 (=131072) | tile_dim0=256
  g1[4] = 16;                   // tile_dim1 = 16 rows
  g1[5] = 256;                  // tensor_dim0_stride = 256 elements
  g1[6] = 0;
  g1[7] = 0;
  v4i z4 = {};
#if defined(__clang_major__) && (__clang_major__ >= 23)
  v8i z8 = {};
  __builtin_amdgcn_tensor_load_to_lds(g0, g1, z4, z4, z8, 0);
#else
  __builtin_amdgcn_tensor_load_to_lds(g0, g1, z4, z4, 0);
#endif
}

// ---------------------------------------------------------------------------
// Prep: dts + re_bias
// ---------------------------------------------------------------------------
__global__ void tcfc_prep(const float* __restrict__ sensor_time,
                          const float* __restrict__ re_norm,
                          const float* __restrict__ re_W,
                          const float* __restrict__ re_b,
                          float* __restrict__ dts, float* __restrict__ re_bias) {
  int i = threadIdx.x;
  if (i < T_STEPS)
    dts[i] = (i == 0) ? sensor_time[0] : sensor_time[i] - sensor_time[i - 1];
  if (i < DMODEL) re_bias[i] = re_norm[0] * re_W[i] + re_b[i];
}

// Convert f32 [rows][cols] -> f16 transposed [cols][rows]
__global__ void tcfc_cvtT(const float* __restrict__ src, h16_t* __restrict__ dst,
                          int rows, int cols) {
  int i = blockIdx.x * blockDim.x + threadIdx.x;
  if (i >= rows * cols) return;
  int r = i / cols, c = i % cols;
  dst[(size_t)c * rows + r] = (h16_t)src[i];
}

// ---------------------------------------------------------------------------
// LN1 + QKV GEMM. 16 rows per block (same t). Q,K -> qk [T*K][512];
// V -> vT [T][H][DH][K].
// ---------------------------------------------------------------------------
__global__ __launch_bounds__(128) void tcfc_ln_qkv(
    const float* __restrict__ x, const float* __restrict__ g,
    const float* __restrict__ b, const h16_t* __restrict__ wT,
    const float* __restrict__ bias, h16_t* __restrict__ qk,
    h16_t* __restrict__ vT) {
  const int LDH = DMODEL + 8;
  __shared__ h16_t h16[16 * LDH];
  __shared__ float s_sum[16][8], s_sq[16][8], s_mu[16], s_rs[16];
  int tid = threadIdx.x;
  int rowbase = blockIdx.x * 16;
  int row = tid >> 3, seg = tid & 7;
  const float* xr = x + (size_t)(rowbase + row) * DMODEL + seg * 32;
  float xv[32];
  float s = 0.f, q = 0.f;
#pragma unroll
  for (int i = 0; i < 32; ++i) { float v = xr[i]; xv[i] = v; s += v; q += v * v; }
  s_sum[row][seg] = s; s_sq[row][seg] = q;
  __syncthreads();
  if (tid < 16) {
    float ss = 0.f, qq = 0.f;
#pragma unroll
    for (int j = 0; j < 8; ++j) { ss += s_sum[tid][j]; qq += s_sq[tid][j]; }
    float mu = ss / DMODEL;
    float var = qq / DMODEL - mu * mu;
    s_mu[tid] = mu; s_rs[tid] = rsqrtf(var + 1e-5f);
  }
  __syncthreads();
  {
    float mu = s_mu[row], rs = s_rs[row];
    int c0 = seg * 32;
#pragma unroll
    for (int i = 0; i < 32; ++i) {
      int c = c0 + i;
      h16[row * LDH + c] = (h16_t)((xv[i] - mu) * rs * g[c] + b[c]);
    }
  }
  __syncthreads();
  int wv = tid >> 5, lane = tid & 31, colid = lane & 15, rh = lane >> 4;
  int t = rowbase >> 8, tok0 = rowbase & 255;
  v16h af[8];
#pragma unroll
  for (int kk = 0; kk < 8; ++kk) af[kk] = load_a_frag(h16, LDH, kk * 32);
  // 12 tiles per wave, processed in pairs so tileB loads overlap tileA WMMAs.
  for (int tp = 0; tp < 6; ++tp) {
    int n0A = (wv * 12 + tp * 2) * 16;
    const h16_t* wpA = wT + (size_t)n0A * DMODEL;
    const h16_t* wpB = wpA + (size_t)16 * DMODEL;
    v16h bfA[8], bfB[8];
#pragma unroll
    for (int kk = 0; kk < 8; ++kk) bfA[kk] = load_bT_frag(wpA, DMODEL, kk * 32);
#pragma unroll
    for (int kk = 0; kk < 8; ++kk) bfB[kk] = load_bT_frag(wpB, DMODEL, kk * 32);
    v8f accA = {}, accB = {};
#pragma unroll
    for (int kk = 0; kk < 8; ++kk) accA = wmma_f16(af[kk], bfA[kk], accA);
#pragma unroll
    for (int kk = 0; kk < 8; ++kk) accB = wmma_f16(af[kk], bfB[kk], accB);
    SGB(0x020, 32);  // all 32 b128 B loads first (one clause, both tiles)
    SGB(0x008, 16);  // then 16 WMMAs (tileA's run while tileB loads land)
    v8f accs[2] = {accA, accB};
#pragma unroll
    for (int p = 0; p < 2; ++p) {
      int col = n0A + p * 16 + colid;
      float bs = bias[col];
      if (col < 512) {
#pragma unroll
        for (int j = 0; j < 8; ++j)
          qk[(size_t)(rowbase + rh * 8 + j) * 512 + col] =
              (h16_t)(accs[p][j] + bs);
      } else {
        int hh = (col - 512) >> 6, dh = (col - 512) & 63;
        h16_t* vp = vT + ((size_t)(t * NHEAD + hh) * DHEAD + dh) * K_TOK;
#pragma unroll
        for (int j = 0; j < 8; ++j)
          vp[tok0 + rh * 8 + j] = (h16_t)(accs[p][j] + bs);
      }
    }
  }
}

// ---------------------------------------------------------------------------
// Flash attention per (t, head). 8 waves x 2 query tiles of 16 rows.
// ---------------------------------------------------------------------------
__global__ __launch_bounds__(256) void tcfc_attn(const h16_t* __restrict__ qk,
                                                 const h16_t* __restrict__ vT,
                                                 h16_t* __restrict__ o) {
  const int LDP = 40;  // padded P-tile row stride (halfs)
  __shared__ h16_t pbuf[8][16 * LDP];
  int t = blockIdx.x >> 2, hh = blockIdx.x & 3;
  int wv = threadIdx.x >> 5, lane = threadIdx.x & 31;
  int colid = lane & 15, rh = lane >> 4;
  const h16_t* qbase = qk + (size_t)t * 256 * 512 + hh * 64;
  const h16_t* kbase = qbase + 256;
  const h16_t* vbase = vT + (size_t)(t * NHEAD + hh) * DHEAD * K_TOK;
  h16_t* pb = pbuf[wv];
  for (int qi = 0; qi < 2; ++qi) {
    int qt = wv + qi * 8;
    const h16_t* qp = qbase + (size_t)qt * 16 * 512;
    v16h aq0 = load_a_frag(qp, 512, 0);
    v16h aq1 = load_a_frag(qp, 512, 32);
    v8f oa0 = {}, oa1 = {}, oa2 = {}, oa3 = {};
    float m[8], l[8];
#pragma unroll
    for (int j = 0; j < 8; ++j) { m[j] = -1e30f; l[j] = 0.f; }
    for (int kc = 0; kc < 8; ++kc) {
      int n0 = kc * 32;
      v16h bk0 = load_bT_frag(kbase + (size_t)n0 * 512, 512, 0);
      v16h bk1 = load_bT_frag(kbase + (size_t)n0 * 512, 512, 32);
      v16h bk2 = load_bT_frag(kbase + (size_t)(n0 + 16) * 512, 512, 0);
      v16h bk3 = load_bT_frag(kbase + (size_t)(n0 + 16) * 512, 512, 32);
      // V fragment loads issued early: independent of softmax math below.
      v16h bv0 = load_bT_frag(vbase + 0 * 16 * K_TOK, K_TOK, n0);
      v16h bv1 = load_bT_frag(vbase + 1 * 16 * K_TOK, K_TOK, n0);
      v16h bv2 = load_bT_frag(vbase + 2 * 16 * K_TOK, K_TOK, n0);
      v16h bv3 = load_bT_frag(vbase + 3 * 16 * K_TOK, K_TOK, n0);
      v8f sa = {}, sb = {};
      sa = wmma_f16(aq0, bk0, sa);
      sa = wmma_f16(aq1, bk1, sa);
      sb = wmma_f16(aq0, bk2, sb);
      sb = wmma_f16(aq1, bk3, sb);
#pragma unroll
      for (int j = 0; j < 8; ++j) { sa[j] *= 0.125f; sb[j] *= 0.125f; }
      float cm[8];
#pragma unroll
      for (int j = 0; j < 8; ++j) cm[j] = fmaxf(sa[j], sb[j]);
#pragma unroll
      for (int off = 1; off < 16; off <<= 1)
#pragma unroll
        for (int j = 0; j < 8; ++j) cm[j] = fmaxf(cm[j], __shfl_xor(cm[j], off, 32));
      float pa[8], pv[8], alpha[8];
#pragma unroll
      for (int j = 0; j < 8; ++j) {
        float mn = fmaxf(m[j], cm[j]);
        alpha[j] = __expf(m[j] - mn);
        pa[j] = __expf(sa[j] - mn);
        pv[j] = __expf(sb[j] - mn);
        m[j] = mn;
      }
      float rs[8];
#pragma unroll
      for (int j = 0; j < 8; ++j) rs[j] = pa[j] + pv[j];
#pragma unroll
      for (int off = 1; off < 16; off <<= 1)
#pragma unroll
        for (int j = 0; j < 8; ++j) rs[j] += __shfl_xor(rs[j], off, 32);
#pragma unroll
      for (int j = 0; j < 8; ++j) {
        l[j] = l[j] * alpha[j] + rs[j];
        oa0[j] *= alpha[j]; oa1[j] *= alpha[j];
        oa2[j] *= alpha[j]; oa3[j] *= alpha[j];
      }
#pragma unroll
      for (int j = 0; j < 8; ++j) {
        pb[(rh * 8 + j) * LDP + colid] = (h16_t)pa[j];
        pb[(rh * 8 + j) * LDP + 16 + colid] = (h16_t)pv[j];
      }
      asm volatile("s_wait_dscnt 0" ::: "memory");
      v16h ap = load_a_frag(pb, LDP, 0);
      oa0 = wmma_f16(ap, bv0, oa0);
      oa1 = wmma_f16(ap, bv1, oa1);
      oa2 = wmma_f16(ap, bv2, oa2);
      oa3 = wmma_f16(ap, bv3, oa3);
      // Pipeline spec: K loads, V loads (overlap S WMMAs), S WMMAs, softmax
      // VALU floats free, P stores, P reload, PV WMMAs.
      SGB(0x020, 8);   // K fragment loads
      SGB(0x020, 8);   // V fragment loads (in flight during S WMMAs)
      SGB(0x008, 4);   // S = Q*K^T WMMAs
      SGB(0x200, 16);  // P staging stores
      SGB(0x100, 2);   // P A-fragment reload
      SGB(0x008, 4);   // O += P*V WMMAs
    }
    float inv[8];
#pragma unroll
    for (int j = 0; j < 8; ++j) inv[j] = 1.f / l[j];
    h16_t* ob = o + ((size_t)t * 256 + qt * 16) * DMODEL + hh * 64;
#pragma unroll
    for (int j = 0; j < 8; ++j) {
      size_t r = (size_t)(rh * 8 + j) * DMODEL;
      ob[r + 0 * 16 + colid] = (h16_t)(oa0[j] * inv[j]);
      ob[r + 1 * 16 + colid] = (h16_t)(oa1[j] * inv[j]);
      ob[r + 2 * 16 + colid] = (h16_t)(oa2[j] * inv[j]);
      ob[r + 3 * 16 + colid] = (h16_t)(oa3[j] * inv[j]);
    }
  }
}

// ---------------------------------------------------------------------------
// out = resid + o @ out_W + out_b
// ---------------------------------------------------------------------------
__global__ __launch_bounds__(128) void tcfc_out_proj(
    const h16_t* __restrict__ o, const h16_t* __restrict__ woT,
    const float* __restrict__ ob, const float* __restrict__ resid,
    float* __restrict__ out) {
  int tid = threadIdx.x;
  int rowbase = blockIdx.x * 16;
  int wv = tid >> 5, lane = tid & 31, colid = lane & 15, rh = lane >> 4;
  const h16_t* arow = o + (size_t)rowbase * DMODEL;
  v16h af[8];
#pragma unroll
  for (int kk = 0; kk < 8; ++kk) af[kk] = load_a_frag(arow, DMODEL, kk * 32);
  for (int tp = 0; tp < 2; ++tp) {
    int n0A = (wv * 4 + tp * 2) * 16;
    const h16_t* wpA = woT + (size_t)n0A * DMODEL;
    const h16_t* wpB = wpA + (size_t)16 * DMODEL;
    v16h bfA[8], bfB[8];
#pragma unroll
    for (int kk = 0; kk < 8; ++kk) bfA[kk] = load_bT_frag(wpA, DMODEL, kk * 32);
#pragma unroll
    for (int kk = 0; kk < 8; ++kk) bfB[kk] = load_bT_frag(wpB, DMODEL, kk * 32);
    v8f accA = {}, accB = {};
#pragma unroll
    for (int kk = 0; kk < 8; ++kk) accA = wmma_f16(af[kk], bfA[kk], accA);
#pragma unroll
    for (int kk = 0; kk < 8; ++kk) accB = wmma_f16(af[kk], bfB[kk], accB);
    SGB(0x020, 32);
    SGB(0x008, 16);
    v8f accs[2] = {accA, accB};
#pragma unroll
    for (int p = 0; p < 2; ++p) {
      int col = n0A + p * 16 + colid;
      float bs = ob[col];
#pragma unroll
      for (int j = 0; j < 8; ++j) {
        size_t idx = (size_t)(rowbase + rh * 8 + j) * DMODEL + col;
        out[idx] = resid[idx] + accs[p][j] + bs;
      }
    }
  }
}

// ---------------------------------------------------------------------------
// LN2 + MLP (gelu) + residual + re_bias, emitting the CfC input as f16.
// ---------------------------------------------------------------------------
__global__ __launch_bounds__(128) void tcfc_ln_mlp(
    const float* __restrict__ x, const float* __restrict__ g,
    const float* __restrict__ b, const h16_t* __restrict__ w1T,
    const float* __restrict__ b1, const h16_t* __restrict__ w2T,
    const float* __restrict__ b2, const float* __restrict__ re_bias,
    h16_t* __restrict__ out16) {
  const int LDH = DMODEL + 8;
  const int LDG = 1024 + 8;
  __shared__ h16_t h16[16 * LDH];
  __shared__ h16_t g16[16 * LDG];
  __shared__ float s_sum[16][8], s_sq[16][8], s_mu[16], s_rs[16];
  int tid = threadIdx.x;
  int rowbase = blockIdx.x * 16;
  int row = tid >> 3, seg = tid & 7;
  const float* xr = x + (size_t)(rowbase + row) * DMODEL + seg * 32;
  float xv[32];
  float s = 0.f, q = 0.f;
#pragma unroll
  for (int i = 0; i < 32; ++i) { float v = xr[i]; xv[i] = v; s += v; q += v * v; }
  s_sum[row][seg] = s; s_sq[row][seg] = q;
  __syncthreads();
  if (tid < 16) {
    float ss = 0.f, qq = 0.f;
#pragma unroll
    for (int j = 0; j < 8; ++j) { ss += s_sum[tid][j]; qq += s_sq[tid][j]; }
    float mu = ss / DMODEL;
    float var = qq / DMODEL - mu * mu;
    s_mu[tid] = mu; s_rs[tid] = rsqrtf(var + 1e-5f);
  }
  __syncthreads();
  {
    float mu = s_mu[row], rs = s_rs[row];
    int c0 = seg * 32;
#pragma unroll
    for (int i = 0; i < 32; ++i) {
      int c = c0 + i;
      h16[row * LDH + c] = (h16_t)((xv[i] - mu) * rs * g[c] + b[c]);
    }
  }
  __syncthreads();
  int wv = tid >> 5, lane = tid & 31, colid = lane & 15, rh = lane >> 4;
  v16h af[8];
#pragma unroll
  for (int kk = 0; kk < 8; ++kk) af[kk] = load_a_frag(h16, LDH, kk * 32);
  // GEMM1: 16 tiles per wave, processed in pairs.
  for (int tp = 0; tp < 8; ++tp) {
    int n0A = (wv * 16 + tp * 2) * 16;
    const h16_t* wpA = w1T + (size_t)n0A * DMODEL;
    const h16_t* wpB = wpA + (size_t)16 * DMODEL;
    v16h bfA[8], bfB[8];
#pragma unroll
    for (int kk = 0; kk < 8; ++kk) bfA[kk] = load_bT_frag(wpA, DMODEL, kk * 32);
#pragma unroll
    for (int kk = 0; kk < 8; ++kk) bfB[kk] = load_bT_frag(wpB, DMODEL, kk * 32);
    v8f accA = {}, accB = {};
#pragma unroll
    for (int kk = 0; kk < 8; ++kk) accA = wmma_f16(af[kk], bfA[kk], accA);
#pragma unroll
    for (int kk = 0; kk < 8; ++kk) accB = wmma_f16(af[kk], bfB[kk], accB);
    SGB(0x020, 32);
    SGB(0x008, 16);
    v8f accs[2] = {accA, accB};
#pragma unroll
    for (int p = 0; p < 2; ++p) {
      int col = n0A + p * 16 + colid;
      float bs = b1[col];
#pragma unroll
      for (int j = 0; j < 8; ++j) {
        float v = accs[p][j] + bs;
        float t3 = v + 0.044715f * v * v * v;
        float gl = 0.5f * v * (1.f + tanhf(0.7978845608f * t3));
        g16[(rh * 8 + j) * LDG + col] = (h16_t)gl;
      }
    }
  }
  __syncthreads();
  for (int tile = wv * 4; tile < wv * 4 + 4; ++tile) {
    int n0 = tile * 16;
    const h16_t* wp = w2T + (size_t)n0 * 1024;
    v8f acc = {};
#pragma unroll
    for (int ch = 0; ch < 4; ++ch) {
      v16h a2[8], bf[8];
#pragma unroll
      for (int kk = 0; kk < 8; ++kk) {
        int kq = (ch * 8 + kk) * 32;
        a2[kk] = load_a_frag(g16, LDG, kq);
        bf[kk] = load_bT_frag(wp, 1024, kq);
      }
#pragma unroll
      for (int kk = 0; kk < 8; ++kk) acc = wmma_f16(a2[kk], bf[kk], acc);
      SGB(0x020, 16);
      SGB(0x008, 8);
    }
    int col = n0 + colid;
    float bs = b2[col] + re_bias[col];
#pragma unroll
    for (int j = 0; j < 8; ++j) {
      size_t idx = (size_t)(rowbase + rh * 8 + j) * DMODEL + col;
      out16[idx] = (h16_t)(x[idx] + acc[j] + bs);
    }
  }
}

// ---------------------------------------------------------------------------
// CfC scan: K rows are independent recurrences; only T is sequential.
// x_t tile fetched via TDM tensor_load_to_lds; h ping-pong in LDS; two
// pipelined WMMA passes per step with explicit pipeline scheduling spec.
// ---------------------------------------------------------------------------
__global__ __launch_bounds__(256) void tcfc_cfc_scan(
    const h16_t* __restrict__ x16g, const float* __restrict__ dts,
    const h16_t* __restrict__ w1T, const h16_t* __restrict__ w2T,
    const h16_t* __restrict__ waT, const h16_t* __restrict__ wbT,
    const float* __restrict__ b1, const float* __restrict__ b2,
    const float* __restrict__ ba, const float* __restrict__ bb,
    const float* __restrict__ prev, float* __restrict__ out) {
  const int LDX = DMODEL + 8;  // must match TDM pad: 512B row + 16B pad
  __shared__ h16_t x16[16 * LDX];
  __shared__ h16_t hbuf[2][16 * LDX];
  int tid = threadIdx.x;
  int k0 = blockIdx.x * 16;
  for (int i = tid; i < 16 * LDX; i += 256) {
    hbuf[0][i] = (h16_t)0.f;
    hbuf[1][i] = (h16_t)0.f;
  }
  int wv = tid >> 5, lane = tid & 31, colid = lane & 15, rh = lane >> 4;
  __syncthreads();
  uint32_t lds_x = (uint32_t)(uintptr_t)(&x16[0]);
  for (int t = 0; t < T_STEPS; ++t) {
    if (wv == 0) {
      const h16_t* xp = x16g + ((size_t)t * K_TOK + k0) * DMODEL;
      tdm_load_tile16x256_f16(lds_x, (uint64_t)(uintptr_t)xp);
      __builtin_amdgcn_s_wait_tensorcnt((short)0);
    }
    __syncthreads();
    int cur = t & 1, nxt = cur ^ 1;
    float dt = dts[t];
    const h16_t* hc = hbuf[cur];
    int n0 = (wv * 2) * 16;
    for (int it = 0; it < 2; ++it, n0 += 16) {
      const h16_t* p1 = w1T + (size_t)n0 * 512;
      const h16_t* p2 = w2T + (size_t)n0 * 512;
      const h16_t* pa = waT + (size_t)n0 * 512;
      const h16_t* pw = wbT + (size_t)n0 * 512;
      int col = n0 + colid;
      // ---- pass 1: W1, W2 (2-deep rotated pipeline) ----
      v8f a1 = {}, a2 = {};
      {
        v16h afc = load_a_frag(x16, LDX, 0);
        v16h b1c = load_bT_frag(p1, 512, 0);
        v16h b2c = load_bT_frag(p2, 512, 0);
#pragma unroll
        for (int kk = 0; kk < 16; ++kk) {
          v16h afn, b1n, b2n;
          if (kk < 15) {
            int kn = kk + 1;
            afn = (kn < 8) ? load_a_frag(x16, LDX, kn * 32)
                           : load_a_frag(hc, LDX, (kn - 8) * 32);
            b1n = load_bT_frag(p1, 512, kn * 32);
            b2n = load_bT_frag(p2, 512, kn * 32);
          }
          a1 = wmma_f16(afc, b1c, a1);
          a2 = wmma_f16(afc, b2c, a2);
          if (kk < 15) { afc = afn; b1c = b1n; b2c = b2n; }
        }
        // pipeline spec: loads(k+1) pinned ahead of WMMAs(k)
        SGB(0x100, 2); SGB(0x020, 4);  // initial fragment loads
#pragma unroll
        for (int kk = 0; kk < 16; ++kk) {
          if (kk < 15) { SGB(0x100, 2); SGB(0x020, 4); }
          SGB(0x008, 2);
        }
      }
      float c1 = b1[col], c2 = b2[col];
      float f1v[8], f2v[8];
#pragma unroll
      for (int j = 0; j < 8; ++j) {
        f1v[j] = tanhf(a1[j] + c1);
        f2v[j] = tanhf(a2[j] + c2);
      }
      // ---- pass 2: Wa, Wb (2-deep rotated pipeline) ----
      v8f aa = {}, ab = {};
      {
        v16h afc = load_a_frag(x16, LDX, 0);
        v16h bac = load_bT_frag(pa, 512, 0);
        v16h bbc = load_bT_frag(pw, 512, 0);
#pragma unroll
        for (int kk = 0; kk < 16; ++kk) {
          v16h afn, ban, bbn;
          if (kk < 15) {
            int kn = kk + 1;
            afn = (kn < 8) ? load_a_frag(x16, LDX, kn * 32)
                           : load_a_frag(hc, LDX, (kn - 8) * 32);
            ban = load_bT_frag(pa, 512, kn * 32);
            bbn = load_bT_frag(pw, 512, kn * 32);
          }
          aa = wmma_f16(afc, bac, aa);
          ab = wmma_f16(afc, bbc, ab);
          if (kk < 15) { afc = afn; bac = ban; bbc = bbn; }
        }
        SGB(0x100, 2); SGB(0x020, 4);
#pragma unroll
        for (int kk = 0; kk < 16; ++kk) {
          if (kk < 15) { SGB(0x100, 2); SGB(0x020, 4); }
          SGB(0x008, 2);
        }
      }
      float ca = ba[col], cb = bb[col];
#pragma unroll
      for (int j = 0; j < 8; ++j) {
        int r = rh * 8 + j;
        float ti = 1.f / (1.f + __expf(-((aa[j] + ca) * dt + (ab[j] + cb))));
        float hn = f1v[j] * (1.f - ti) + ti * f2v[j];
        hbuf[nxt][r * LDX + col] = (h16_t)hn;
        size_t oi = ((size_t)t * K_TOK + k0 + r) * DMODEL + col;
        out[oi] = prev ? prev[oi] + hn : hn;
      }
    }
    __syncthreads();
  }
}

// ---------------------------------------------------------------------------
// Host side
// ---------------------------------------------------------------------------
extern "C" void kernel_launch(void* const* d_in, const int* in_sizes, int n_in,
                              void* d_out, int out_size, void* d_ws,
                              size_t ws_size, hipStream_t stream) {
  (void)in_sizes; (void)n_in; (void)out_size; (void)ws_size;
  const float* spatial = (const float*)d_in[0];
  const float* re_norm = (const float*)d_in[1];
  const float* sensor_time = (const float*)d_in[2];
  const float* re_W = (const float*)d_in[3];
  const float* re_b = (const float*)d_in[4];
  const float* ln1_g = (const float*)d_in[5];
  const float* ln1_b = (const float*)d_in[6];
  const float* qkv_W = (const float*)d_in[7];
  const float* qkv_b = (const float*)d_in[8];
  const float* out_W = (const float*)d_in[9];
  const float* out_b = (const float*)d_in[10];
  const float* ln2_g = (const float*)d_in[11];
  const float* ln2_b = (const float*)d_in[12];
  const float* mlp1_W = (const float*)d_in[13];
  const float* mlp1_b = (const float*)d_in[14];
  const float* mlp2_W = (const float*)d_in[15];
  const float* mlp2_b = (const float*)d_in[16];
  const float* ff1_W = (const float*)d_in[17];
  const float* ff1_b = (const float*)d_in[18];
  const float* ff2_W = (const float*)d_in[19];
  const float* ff2_b = (const float*)d_in[20];
  const float* ta_W = (const float*)d_in[21];
  const float* ta_b = (const float*)d_in[22];
  const float* tb_W = (const float*)d_in[23];
  const float* tb_b = (const float*)d_in[24];

  uint8_t* ws = (uint8_t*)d_ws;
  size_t off = 0;
  auto alloc = [&](size_t bytes) -> void* {
    void* p = ws + off;
    off += (bytes + 255) & ~(size_t)255;
    return p;
  };

  const size_t NROW = (size_t)T_STEPS * K_TOK;  // 131072
  h16_t* wqT = (h16_t*)alloc(768 * 256 * sizeof(h16_t));
  h16_t* woT = (h16_t*)alloc(256 * 256 * sizeof(h16_t));
  h16_t* wm1T = (h16_t*)alloc(1024 * 256 * sizeof(h16_t));
  h16_t* wm2T = (h16_t*)alloc(256 * 1024 * sizeof(h16_t));
  h16_t* wffT[NLAYER][4];
  for (int l = 0; l < NLAYER; ++l)
    for (int w = 0; w < 4; ++w)
      wffT[l][w] = (h16_t*)alloc(256 * 512 * sizeof(h16_t));
  float* dts = (float*)alloc(T_STEPS * sizeof(float));
  float* rebias = (float*)alloc(DMODEL * sizeof(float));
  h16_t* qk16 = (h16_t*)alloc(NROW * 512 * sizeof(h16_t));
  h16_t* vT16 = (h16_t*)alloc((size_t)T_STEPS * NHEAD * DHEAD * K_TOK * sizeof(h16_t));
  h16_t* o16 = (h16_t*)alloc(NROW * DMODEL * sizeof(h16_t));
  float* abuf = (float*)alloc(NROW * DMODEL * sizeof(float));
  h16_t* x16buf = (h16_t*)alloc(NROW * DMODEL * sizeof(h16_t));
  float* seqbuf = (float*)alloc(NROW * DMODEL * sizeof(float));

  tcfc_prep<<<1, 512, 0, stream>>>(sensor_time, re_norm, re_W, re_b, dts, rebias);

  auto cvt = [&](const float* src, h16_t* dst, int rows, int cols) {
    int n = rows * cols;
    tcfc_cvtT<<<(n + 255) / 256, 256, 0, stream>>>(src, dst, rows, cols);
  };
  cvt(qkv_W, wqT, 256, 768);
  cvt(out_W, woT, 256, 256);
  cvt(mlp1_W, wm1T, 256, 1024);
  cvt(mlp2_W, wm2T, 1024, 256);
  for (int l = 0; l < NLAYER; ++l) {
    cvt(ff1_W + (size_t)l * 512 * 256, wffT[l][0], 512, 256);
    cvt(ff2_W + (size_t)l * 512 * 256, wffT[l][1], 512, 256);
    cvt(ta_W + (size_t)l * 512 * 256, wffT[l][2], 512, 256);
    cvt(tb_W + (size_t)l * 512 * 256, wffT[l][3], 512, 256);
  }

  const int NBLK16 = (int)(NROW / 16);  // 8192
  for (int l = 0; l < NLAYER; ++l) {
    const float* in_seq = (l == 0) ? spatial : seqbuf;
    tcfc_ln_qkv<<<NBLK16, 128, 0, stream>>>(in_seq, ln1_g, ln1_b, wqT, qkv_b,
                                            qk16, vT16);
    tcfc_attn<<<T_STEPS * NHEAD, 256, 0, stream>>>(qk16, vT16, o16);
    tcfc_out_proj<<<NBLK16, 128, 0, stream>>>(o16, woT, out_b, in_seq, abuf);
    tcfc_ln_mlp<<<NBLK16, 128, 0, stream>>>(abuf, ln2_g, ln2_b, wm1T, mlp1_b,
                                            wm2T, mlp2_b, rebias, x16buf);
    float* scan_out = (l == 0) ? seqbuf : (float*)d_out;
    const float* scan_prev = (l == 0) ? nullptr : seqbuf;
    tcfc_cfc_scan<<<K_TOK / 16, 256, 0, stream>>>(
        x16buf, dts, wffT[l][0], wffT[l][1], wffT[l][2], wffT[l][3],
        ff1_b + (size_t)l * DMODEL, ff2_b + (size_t)l * DMODEL,
        ta_b + (size_t)l * DMODEL, tb_b + (size_t)l * DMODEL, scan_prev,
        scan_out);
  }
}
